// QGFDLayer_31104153157928
// MI455X (gfx1250) — matrix operand order repaired and checked by reading, not compiled, not run on hardware
//
#include <hip/hip_runtime.h>

typedef __attribute__((ext_vector_type(16))) _Float16 v16h;
typedef __attribute__((ext_vector_type(8)))  _Float16 v8h;
typedef __attribute__((ext_vector_type(8)))  float    v8f;
typedef int v4i __attribute__((ext_vector_type(4)));
typedef v4i __attribute__((address_space(1))) v4i_g;   // global int4
typedef v4i __attribute__((address_space(3))) v4i_l;   // LDS int4

#define TBM 256
#define TBN 128
#define TBK 32
#define LDS_A 40    // LDS row stride (halves) for A tile and transposed-B tile
#define LDS_B 136   // LDS row stride (halves) for normal B tile

// ---- CDNA5 async global->LDS copy (guarded; falls back to direct LDS store) ----
#if defined(__has_builtin)
#if __has_builtin(__builtin_amdgcn_global_load_async_to_lds_b128)
#define USE_ASYNC_LDS 1
#endif
#endif
#ifndef USE_ASYNC_LDS
#define USE_ASYNC_LDS 0
#endif

__device__ __forceinline__ void copy16_g2l(_Float16* dst, const _Float16* src) {
#if USE_ASYNC_LDS
  __builtin_amdgcn_global_load_async_to_lds_b128((v4i_g*)src, (v4i_l*)dst, 0, 0);
#else
  *(v8h*)dst = *(const v8h*)src;
#endif
}

__device__ __forceinline__ void async_stage_wait() {
#if USE_ASYNC_LDS
#if __has_builtin(__builtin_amdgcn_s_wait_asynccnt)
  __builtin_amdgcn_s_wait_asynccnt(0);
#else
  asm volatile("s_wait_asynccnt 0x0" ::: "memory");
#endif
#endif
}

struct GemmP {
  const _Float16* A; long lda; int aMod; long aSO; long aSI;
  const _Float16* B; long ldb; int bMod; long bSO; long bSI;
  int transB;
  int M; int N; int K;
  float alpha;
  const float* bias;                       // len N, nullable
  const float* C0; long ldc0; long c0S; float beta;  // out += beta*C0
  float* Cf; _Float16* Ch; long ldc; int cMod; long cSO; long cSI;
};

// D = alpha*(A@B) + bias + beta*C0 ; A row-major MxK, B row-major KxN (or
// transB: B supplied NxK row-major). f16 in, f32 accum via
// v_wmma_f32_16x16x32_f16. 256 threads = 8 waves; block tile 256x128; waves
// arranged 4(M)x2(N), each wave computes 64x64 = 4x4 WMMA tiles (16 WMMA /
// K-step). LDS double-buffered; next tile staged asynchronously while the
// current tile is consumed.
__global__ __launch_bounds__(256) void wmma_gemm_f16(GemmP g) {
  __shared__ __align__(32) _Float16 sA[2][TBM * LDS_A];  // 2 x 10240 halves
  __shared__ __align__(32) _Float16 sB[2][TBN * LDS_A];  // 2 x 5120  (>= 32*LDS_B)

  const int tid  = threadIdx.x;
  const int lane = tid & 31;
  const int wave = tid >> 5;
  const int wm = (wave & 3) * 64;   // wave M offset in block tile
  const int wn = (wave >> 2) * 64;  // wave N offset in block tile
  const int batch = blockIdx.z;
  const int m0 = blockIdx.y * TBM;
  const int n0 = blockIdx.x * TBN;

  const _Float16* Ab = g.A + (long)(batch / g.aMod) * g.aSO + (long)(batch % g.aMod) * g.aSI;
  const _Float16* Bb = g.B + (long)(batch / g.bMod) * g.bSO + (long)(batch % g.bMod) * g.bSI;

  v8h zh;
#pragma unroll
  for (int e = 0; e < 8; ++e) zh[e] = (_Float16)0.0f;

  const bool fullA = (m0 + TBM <= g.M);
  const bool fullB = (n0 + TBN <= g.N);

  // ---- tile staging (chunks of 8 halves / 16B per lane) ----
  auto stageA = [&](int buf, int k0) {
    if (fullA) {
#pragma unroll
      for (int it = 0; it < (TBM * TBK / 8) / 256; ++it) {
        int c = tid + it * 256;
        int row = c >> 2, col = (c & 3) * 8;
        copy16_g2l(&sA[buf][row * LDS_A + col],
                   Ab + (long)(m0 + row) * g.lda + (k0 + col));
      }
    } else {
#pragma unroll
      for (int it = 0; it < (TBM * TBK / 8) / 256; ++it) {
        int c = tid + it * 256;
        int row = c >> 2, col = (c & 3) * 8;
        int gm = m0 + row;
        v8h v = zh;
        if (gm < g.M) v = *(const v8h*)(Ab + (long)gm * g.lda + (k0 + col));
        *(v8h*)&sA[buf][row * LDS_A + col] = v;
      }
    }
  };
  auto stageB = [&](int buf, int k0) {
    if (!g.transB) {            // B is KxN row-major: tile TBK x TBN
      if (fullB) {
#pragma unroll
        for (int it = 0; it < (TBK * TBN / 8) / 256; ++it) {
          int c = tid + it * 256;
          int row = c >> 4, col = (c & 15) * 8;
          copy16_g2l(&sB[buf][row * LDS_B + col],
                     Bb + (long)(k0 + row) * g.ldb + (n0 + col));
        }
      } else {
#pragma unroll
        for (int it = 0; it < (TBK * TBN / 8) / 256; ++it) {
          int c = tid + it * 256;
          int row = c >> 4, col = (c & 15) * 8;
          int gn = n0 + col;
          v8h v = zh;
          if (gn < g.N) v = *(const v8h*)(Bb + (long)(k0 + row) * g.ldb + gn);
          *(v8h*)&sB[buf][row * LDS_B + col] = v;
        }
      }
    } else {                    // B supplied NxK row-major: tile TBN x TBK
      if (fullB) {
#pragma unroll
        for (int it = 0; it < (TBN * TBK / 8) / 256; ++it) {
          int c = tid + it * 256;
          int row = c >> 2, col = (c & 3) * 8;
          copy16_g2l(&sB[buf][row * LDS_A + col],
                     Bb + (long)(n0 + row) * g.ldb + (k0 + col));
        }
      } else {
#pragma unroll
        for (int it = 0; it < (TBN * TBK / 8) / 256; ++it) {
          int c = tid + it * 256;
          int row = c >> 2, col = (c & 3) * 8;
          int gn = n0 + row;
          v8h v = zh;
          if (gn < g.N) v = *(const v8h*)(Bb + (long)gn * g.ldb + (k0 + col));
          *(v8h*)&sB[buf][row * LDS_A + col] = v;
        }
      }
    }
  };

  v8f acc[4][4];
#pragma unroll
  for (int i = 0; i < 4; ++i)
#pragma unroll
    for (int j = 0; j < 4; ++j)
#pragma unroll
      for (int e = 0; e < 8; ++e) acc[i][j][e] = 0.0f;

  int buf = 0;
  stageA(0, 0);
  stageB(0, 0);

  for (int k0 = 0; k0 < g.K; k0 += TBK) {
    async_stage_wait();     // this wave's staged tile is in LDS
    __syncthreads();        // everyone's is
    if (k0 + TBK < g.K) {   // kick off next tile's DMA, overlapped with WMMAs
      stageA(buf ^ 1, k0 + TBK);
      stageB(buf ^ 1, k0 + TBK);
    }

    // ---- A fragments (ISA 16-bit A 16x32: lane<16 -> M=lane, K in
    //      {0..7,16..23}; lane>=16 -> M=lane-16, K in {8..15,24..31}) ----
    v16h af[4];
    {
      int mrow = lane & 15;
      int kb = (lane >> 4) * 8;
#pragma unroll
      for (int i = 0; i < 4; ++i) {
        const _Float16* rp = &sA[buf][(wm + i * 16 + mrow) * LDS_A];
        v8h lo = *(const v8h*)(rp + kb);
        v8h hi = *(const v8h*)(rp + kb + 16);
        v16h a;
#pragma unroll
        for (int e = 0; e < 8; ++e) { a[e] = lo[e]; a[8 + e] = hi[e]; }
        af[i] = a;
      }
    }
    // ---- B fragments (ISA 16-bit B 32x16: lane%16 = N col; regs = 16 K vals,
    //      same K order as A), one at a time to bound register pressure ----
#pragma unroll
    for (int j = 0; j < 4; ++j) {
      v16h b;
      if (!g.transB) {
        const _Float16* rp = &sB[buf][wn + j * 16 + (lane & 15)];
#pragma unroll
        for (int e = 0; e < 16; ++e)
          b[e] = rp[((lane >> 4) * 8 + (e & 7) + (e >> 3) * 16) * LDS_B];
      } else {
        const _Float16* rp = &sB[buf][(wn + j * 16 + (lane & 15)) * LDS_A + (lane >> 4) * 8];
        v8h lo = *(const v8h*)(rp);
        v8h hi = *(const v8h*)(rp + 16);
#pragma unroll
        for (int e = 0; e < 8; ++e) { b[e] = lo[e]; b[8 + e] = hi[e]; }
      }
#pragma unroll
      for (int i = 0; i < 4; ++i)
        acc[i][j] = __builtin_amdgcn_wmma_f32_16x16x32_f16(
            false, af[i], false, b, (short)0, acc[i][j], false, false);
    }
    buf ^= 1;
  }

  // ---- epilogue (ISA f32 C/D 16x16: lane = N col; reg r -> M=r (lane<16)
  //      or M=r+8 (lane>=16)) ----
  long cOff = (long)(batch / g.cMod) * g.cSO + (long)(batch % g.cMod) * g.cSI;
  const float* C0b = g.C0 ? (g.C0 + (long)batch * g.c0S) : (const float*)0;
  const int nIdx = lane & 15;
  const int mHalf = (lane >> 4) * 8;
#pragma unroll
  for (int i = 0; i < 4; ++i) {
#pragma unroll
    for (int j = 0; j < 4; ++j) {
      int gn = n0 + wn + j * 16 + nIdx;
      if (gn >= g.N) continue;
      float bv = g.bias ? g.bias[gn] : 0.0f;
#pragma unroll
      for (int r = 0; r < 8; ++r) {
        int gm = m0 + wm + i * 16 + mHalf + r;
        if (gm >= g.M) continue;
        float v = g.alpha * acc[i][j][r] + bv;
        if (C0b) v += g.beta * C0b[(long)gm * g.ldc0 + gn];
        long ci = cOff + (long)gm * g.ldc + gn;
        if (g.Cf) g.Cf[ci] = v;
        if (g.Ch) g.Ch[ci] = (_Float16)v;
      }
    }
  }
}

// Row-wise softmax over `cols` f32 values; writes f16 (always) and f32 (optional,
// may alias the input in place).
__global__ __launch_bounds__(256) void softmax_row(const float* S, long ld,
                                                   float* outF, _Float16* outH,
                                                   int cols) {
  const long row = blockIdx.x;
  const float* src = S + row * ld;
  __shared__ float red[256];
  const int tid = threadIdx.x;

  float mx = -3.402823466e+38f;
  for (int c = tid; c < cols; c += 256) mx = fmaxf(mx, src[c]);
  red[tid] = mx;
  __syncthreads();
  for (int s = 128; s > 0; s >>= 1) {
    if (tid < s) red[tid] = fmaxf(red[tid], red[tid + s]);
    __syncthreads();
  }
  mx = red[0];
  __syncthreads();

  float sum = 0.0f;
  for (int c = tid; c < cols; c += 256) sum += __expf(src[c] - mx);
  red[tid] = sum;
  __syncthreads();
  for (int s = 128; s > 0; s >>= 1) {
    if (tid < s) red[tid] += red[tid + s];
    __syncthreads();
  }
  const float inv = 1.0f / red[0];
  __syncthreads();

  for (int c = tid; c < cols; c += 256) {
    float v = __expf(src[c] - mx) * inv;
    if (outH) outH[row * ld + c] = (_Float16)v;
    if (outF) outF[row * ld + c] = v;
  }
}

__global__ void cvt_f32_to_f16(const float* in, _Float16* out, long n) {
  long i = (long)blockIdx.x * blockDim.x + threadIdx.x;
  if (i < n) out[i] = (_Float16)in[i];
}

extern "C" void kernel_launch(void* const* d_in, const int* in_sizes, int n_in,
                              void* d_out, int out_size, void* d_ws, size_t ws_size,
                              hipStream_t stream) {
  (void)in_sizes; (void)n_in; (void)out_size; (void)ws_size;
  const long Bb = 2, Nn = 2048, Dd = 512, Hh = 8, dh = 64;
  const long ROWS = Bb * Nn;   // 4096
  const long BH = Bb * Hh;     // 16
  const long NN = Nn * Nn;

  const float* X  = (const float*)d_in[0];
  const float* wq = (const float*)d_in[1];
  const float* bq = (const float*)d_in[2];
  const float* wk = (const float*)d_in[3];
  const float* bk = (const float*)d_in[4];
  const float* wv = (const float*)d_in[5];
  const float* bv = (const float*)d_in[6];
  const float* wo = (const float*)d_in[7];
  const float* bo = (const float*)d_in[8];

  float* outO = (float*)d_out;            // [B,N,D] f32
  float* outP = outO + ROWS * Dd;         // [B,H,N,N] f32 -- also scores/sim/p0 scratch

  char* ws = (char*)d_ws;
  size_t off = 0;
  auto take = [&](size_t bytes) -> void* {
    void* p = ws + off;
    off = (off + bytes + 255) & ~(size_t)255;
    return p;
  };
  _Float16* Pf  = (_Float16*)take((size_t)BH * NN * 2);   // transition matrix P (f16)
  _Float16* pA  = (_Float16*)take((size_t)BH * NN * 2);   // p ping
  _Float16* pB  = (_Float16*)take((size_t)BH * NN * 2);   // p pong
  _Float16* Xh  = (_Float16*)take((size_t)ROWS * Dd * 2);
  _Float16* Qh  = (_Float16*)take((size_t)ROWS * Dd * 2); // [B,N,H*64]
  _Float16* Kh  = (_Float16*)take((size_t)ROWS * Dd * 2);
  _Float16* Vh  = (_Float16*)take((size_t)ROWS * Dd * 2);
  _Float16* Wqh = (_Float16*)take((size_t)Dd * Dd * 2);
  _Float16* Wkh = (_Float16*)take((size_t)Dd * Dd * 2);
  _Float16* Wvh = (_Float16*)take((size_t)Dd * Dd * 2);
  _Float16* Woh = (_Float16*)take((size_t)Dd * Dd * 2);
  _Float16* Ah  = (_Float16*)take((size_t)ROWS * Dd * 2); // attn output [B,N,H*64]

  // ---- f32 -> f16 conversions ----
  {
    long n = ROWS * Dd;
    cvt_f32_to_f16<<<dim3((unsigned)((n + 255) / 256)), 256, 0, stream>>>(X, Xh, n);
    long w = Dd * Dd;
    dim3 gw((unsigned)((w + 255) / 256));
    cvt_f32_to_f16<<<gw, 256, 0, stream>>>(wq, Wqh, w);
    cvt_f32_to_f16<<<gw, 256, 0, stream>>>(wk, Wkh, w);
    cvt_f32_to_f16<<<gw, 256, 0, stream>>>(wv, Wvh, w);
    cvt_f32_to_f16<<<gw, 256, 0, stream>>>(wo, Woh, w);
  }

  auto gemm = [&](const _Float16* A, long lda, int aMod, long aSO, long aSI,
                  const _Float16* Bp, long ldb, int bMod, long bSO, long bSI,
                  int transB, int M, int N, int K, float alpha,
                  const float* bias,
                  const float* C0, long ldc0, long c0S, float beta,
                  float* Cf, _Float16* Ch, long ldc, int cMod, long cSO, long cSI,
                  int batches) {
    GemmP g{};
    g.A = A; g.lda = lda; g.aMod = aMod; g.aSO = aSO; g.aSI = aSI;
    g.B = Bp; g.ldb = ldb; g.bMod = bMod; g.bSO = bSO; g.bSI = bSI;
    g.transB = transB; g.M = M; g.N = N; g.K = K; g.alpha = alpha;
    g.bias = bias;
    g.C0 = C0; g.ldc0 = ldc0; g.c0S = c0S; g.beta = beta;
    g.Cf = Cf; g.Ch = Ch; g.ldc = ldc; g.cMod = cMod; g.cSO = cSO; g.cSI = cSI;
    dim3 grid((unsigned)((N + TBN - 1) / TBN), (unsigned)((M + TBM - 1) / TBM),
              (unsigned)batches);
    wmma_gemm_f16<<<grid, 256, 0, stream>>>(g);
  };

  // ---- Q, K, V projections: [4096,512] = Xh @ W + b (f16 out) ----
  gemm(Xh, Dd, 1, 0, 0, Wqh, Dd, 1, 0, 0, 0, (int)ROWS, (int)Dd, (int)Dd, 1.0f, bq,
       nullptr, 0, 0, 0.0f, nullptr, Qh, Dd, 1, 0, 0, 1);
  gemm(Xh, Dd, 1, 0, 0, Wkh, Dd, 1, 0, 0, 0, (int)ROWS, (int)Dd, (int)Dd, 1.0f, bk,
       nullptr, 0, 0, 0.0f, nullptr, Kh, Dd, 1, 0, 0, 1);
  gemm(Xh, Dd, 1, 0, 0, Wvh, Dd, 1, 0, 0, 0, (int)ROWS, (int)Dd, (int)Dd, 1.0f, bv,
       nullptr, 0, 0, 0.0f, nullptr, Vh, Dd, 1, 0, 0, 1);

  // ---- sim = K K^T / 8 per (b,h) -> outP scratch; softmax -> P (f16) ----
  gemm(Kh, Dd, (int)Hh, Nn * Dd, dh, Kh, Dd, (int)Hh, Nn * Dd, dh, 1,
       (int)Nn, (int)Nn, (int)dh, 0.125f, nullptr,
       nullptr, 0, 0, 0.0f, outP, nullptr, Nn, 1, NN, 0, (int)BH);
  softmax_row<<<dim3((unsigned)(BH * Nn)), 256, 0, stream>>>(outP, Nn, nullptr, Pf, (int)Nn);

  // ---- scores = Q K^T / (8+1e-8) -> outP; softmax -> p0 f32 (in place) + p f16 ----
  gemm(Qh, Dd, (int)Hh, Nn * Dd, dh, Kh, Dd, (int)Hh, Nn * Dd, dh, 1,
       (int)Nn, (int)Nn, (int)dh, 1.0f / (8.0f + 1e-8f), nullptr,
       nullptr, 0, 0, 0.0f, outP, nullptr, Nn, 1, NN, 0, (int)BH);
  softmax_row<<<dim3((unsigned)(BH * Nn)), 256, 0, stream>>>(outP, Nn, outP, pA, (int)Nn);

  // ---- 4 diffusion steps: p = 0.5*p0 + 0.5 * p @ P ----
  _Float16* pin = pA;
  _Float16* pout = pB;
  for (int s = 0; s < 4; ++s) {
    float* cf = (s == 3) ? outP : nullptr;   // final step writes f32 p to d_out
    gemm(pin, Nn, 1, NN, 0, Pf, Nn, 1, NN, 0, 0,
         (int)Nn, (int)Nn, (int)Nn, 0.5f, nullptr,
         outP, Nn, NN, 0.5f,
         cf, pout, Nn, 1, NN, 0, (int)BH);
    _Float16* t = pin; pin = pout; pout = t;
  }
  // final p (f16) now in `pin`; f32 copy already in d_out p-region.

  // ---- attn = p @ V : [2048,64] per (b,h) -> Ah laid out [B,N,H*64] ----
  gemm(pin, Nn, 1, NN, 0, Vh, Dd, (int)Hh, Nn * Dd, dh, 0,
       (int)Nn, (int)dh, (int)Nn, 1.0f, nullptr,
       nullptr, 0, 0, 0.0f, nullptr, Ah, Dd, (int)Hh, Nn * Dd, dh, (int)BH);

  // ---- out = attn @ wo + bo -> d_out f32 ----
  gemm(Ah, Dd, 1, 0, 0, Woh, Dd, 1, 0, 0, 0, (int)ROWS, (int)Dd, (int)Dd, 1.0f, bo,
       nullptr, 0, 0, 0.0f, outO, nullptr, Dd, 1, 0, 0, 1);
}
